// Attention_16484084482396
// MI455X (gfx1250) — matrix operand order repaired and checked by reading, not compiled, not run on hardware
//
#include <hip/hip_runtime.h>
#include <hip/hip_bf16.h>

typedef __attribute__((ext_vector_type(16))) __bf16 v16bf;
typedef __attribute__((ext_vector_type(8)))  __bf16 v8bf;
typedef __attribute__((ext_vector_type(8)))  float  v8f;

#define BATCH 8
#define NPIX  4096      // 64*64
#define CCH   256
#define CFP   32        // C/8
#define BN    32768     // BATCH*NPIX

// ---------------------------------------------------------------------------
// Kernel 1: fused projections  f = x@Wf+bf, g = x@Wg+bg, h = x@Wh+bh
// One wave computes a 16x16 output tile, K-loop over C=256 (8 WMMAs).
// f,g stored row-major [BN][32] bf16; h stored transposed [B][C][N] bf16.
// ---------------------------------------------------------------------------
__global__ __launch_bounds__(32) void proj_kernel(
    const float* __restrict__ x,
    const float* __restrict__ wf, const float* __restrict__ wg,
    const float* __restrict__ wh,
    const float* __restrict__ bf, const float* __restrict__ bg,
    const float* __restrict__ bh,
    __bf16* __restrict__ fbuf, __bf16* __restrict__ gbuf,
    __bf16* __restrict__ hT)
{
    const int lane = threadIdx.x & 31;
    const int hi   = lane >> 4;         // half-wave select
    const int ln   = lane & 15;
    const int row0 = blockIdx.x * 16;   // global pixel row tile
    const int ct   = blockIdx.y;        // 0..19 column tiles (f:2, g:2, h:16)

    const float* w; const float* bias; int col0; int wld; int which;
    if (ct < 2)      { w = wf; bias = bf; col0 = ct * 16;       wld = CFP; which = 0; }
    else if (ct < 4) { w = wg; bias = bg; col0 = (ct - 2) * 16; wld = CFP; which = 1; }
    else             { w = wh; bias = bh; col0 = (ct - 4) * 16; wld = CCH; which = 2; }

    v8f acc;
    #pragma unroll
    for (int j = 0; j < 8; j++) acc[j] = 0.0f;

    const int arow = row0 + ln;         // A-matrix row for this lane
    const int aoff = hi * 8;            // K sub-offset per ISA A layout

    for (int k0 = 0; k0 < CCH; k0 += 32) {
        // ---- A tile: x[arow][k0 + {aoff..aoff+7, 16+aoff..16+aoff+7}] f32->bf16
        const float* xp = x + (size_t)arow * CCH + k0 + aoff;
        float af[16];
        *(float4*)(af + 0)  = *(const float4*)(xp);
        *(float4*)(af + 4)  = *(const float4*)(xp + 4);
        *(float4*)(af + 8)  = *(const float4*)(xp + 16);
        *(float4*)(af + 12) = *(const float4*)(xp + 20);
        v16bf a;
        #pragma unroll
        for (int i = 0; i < 16; i++) a[i] = (__bf16)af[i];

        // ---- B tile: w[k0 + {aoff..aoff+7, 16+aoff..16+aoff+7}][col0 + ln]  (lane = N column)
        const int kbase = k0 + hi * 8;
        v16bf bm;
        #pragma unroll
        for (int i = 0; i < 16; i++)
            bm[i] = (__bf16)w[(size_t)(kbase + i + (i & 8)) * wld + col0 + ln];

        acc = __builtin_amdgcn_wmma_f32_16x16x32_bf16(
                  false, a, false, bm, (short)0, acc, false, false);
    }

    const float bv = bias[col0 + ln];
    #pragma unroll
    for (int j = 0; j < 8; j++) acc[j] += bv;

    if (which < 2) {
        __bf16* ob = (which == 0) ? fbuf : gbuf;
        #pragma unroll
        for (int j = 0; j < 8; j++) {
            int m = j + 8 * hi;  // C layout: VGPR j -> row j (+8 for hi lanes)
            ob[(size_t)(row0 + m) * CFP + col0 + ln] = (__bf16)acc[j];
        }
    } else {
        // transposed store: hT[b][ch][pixel]; lane owns one channel, 8 pixels
        const int b  = row0 / NPIX;
        const int n0 = row0 - b * NPIX;
        v8bf o8;
        #pragma unroll
        for (int j = 0; j < 8; j++) o8[j] = (__bf16)acc[j];
        __bf16* dst = hT + ((size_t)b * CCH + col0 + ln) * NPIX + n0 + 8 * hi;
        *(v8bf*)dst = o8;   // packed global_store_b128
    }
}

// ---------------------------------------------------------------------------
// Kernel 2: flash attention.  block = 2 waves, one 16-query tile per block.
// Waves ALTERNATE as softmax producer per 64-key block: producer runs the
// 4 s-WMMAs + online softmax and publishes p / scale / psum to double-
// buffered LDS (one workgroup barrier per iteration); both waves then run
// their 16 o-WMMAs (8 channel tiles each) against h^T.  This halves the
// dominant softmax VALU cost and removes the per-wave p-repack.
// ---------------------------------------------------------------------------
__global__ __launch_bounds__(64) void attn_kernel(
    const float* __restrict__ x, const __bf16* __restrict__ fbuf,
    const __bf16* __restrict__ gbuf, const __bf16* __restrict__ hT,
    const float* __restrict__ gamma, float* __restrict__ out)
{
    __shared__ __attribute__((aligned(32))) __bf16 pbuf[2][16 * 64];
    __shared__ float sscale[2][16];   // exp(m_old - m_new) per row
    __shared__ float spsum[2][16];    // row sum of new p per row
    __shared__ float mstate[16];      // running row max (producer-owned)

    const int tid  = threadIdx.x;
    const int wave = tid >> 5;
    const int lane = tid & 31;
    const int hi   = lane >> 4;
    const int ln   = lane & 15;
    const int b    = blockIdx.y;
    const int q0   = blockIdx.x * 16;
    const size_t rowbase = (size_t)b * NPIX;

    if (tid < 16) mstate[tid] = -3.0e38f;
    __syncthreads();

    // ---- load g A-matrix tile (16 queries x 32 cf); both waves produce
    const __bf16* gp = gbuf + (rowbase + q0 + ln) * CFP + hi * 16;
    v8bf g_lo = *(const v8bf*)(gp);
    v8bf g_hi = *(const v8bf*)(gp + 8);
    v16bf gA;
    #pragma unroll
    for (int i = 0; i < 8; i++) { gA[i] = g_lo[i]; gA[i + 8] = g_hi[i]; }

    v8f acc[8];
    #pragma unroll
    for (int t = 0; t < 8; t++)
        #pragma unroll
        for (int j = 0; j < 8; j++) acc[t][j] = 0.0f;

    float lrow[8];
    #pragma unroll
    for (int j = 0; j < 8; j++) lrow[j] = 0.0f;

    const int c0 = wave * 128;                 // this wave's 128-channel slab
    const __bf16* fB_base = fbuf + rowbase * CFP;
    const __bf16* hT_b    = hT + (size_t)b * CCH * NPIX;

    for (int it = 0; it < NPIX / 64; ++it) {
        const int kb   = it * 64;
        const int bufi = it & 1;

        if (wave == bufi) {
            // ===== producer: s = g @ f^T for 64 keys, online softmax ========
            v8f s[4];
            #pragma unroll
            for (int u = 0; u < 4; u++) {
                const __bf16* fp =
                    fB_base + (size_t)(kb + 16 * u + ln) * CFP + hi * 16;
                v16bf fB = *(const v16bf*)fp;
                v8f z;
                #pragma unroll
                for (int j = 0; j < 8; j++) z[j] = 0.0f;
                s[u] = __builtin_amdgcn_wmma_f32_16x16x32_bf16(
                           false, gA, false, fB, (short)0, z, false, false);
            }
            __bf16* pw = pbuf[bufi];
            #pragma unroll
            for (int j = 0; j < 8; j++) {
                const int m = j + 8 * hi;        // row handled by this lane
                float mold = mstate[m];
                float t = fmaxf(fmaxf(s[0][j], s[1][j]),
                                fmaxf(s[2][j], s[3][j]));
                #pragma unroll
                for (int off = 8; off >= 1; off >>= 1)
                    t = fmaxf(t, __shfl_xor(t, off, 16));
                float nm = fmaxf(mold, t);
                float ps = 0.0f;
                #pragma unroll
                for (int u = 0; u < 4; u++) {
                    float p = __expf(s[u][j] - nm);
                    pw[m * 64 + 16 * u + ln] = (__bf16)p;
                    ps += p;
                }
                #pragma unroll
                for (int off = 8; off >= 1; off >>= 1)
                    ps += __shfl_xor(ps, off, 16);
                sscale[bufi][m] = __expf(mold - nm);
                spsum[bufi][m]  = ps;
                mstate[m]       = nm;
            }
        }
        __syncthreads();   // publish p / scale / psum (s_barrier_signal/wait)

        // ===== consumer (both waves): rescale + o += p @ h ==================
        float scale8[8];
        #pragma unroll
        for (int j = 0; j < 8; j++) {
            const int m = j + 8 * hi;
            scale8[j] = sscale[bufi][m];
            lrow[j]   = lrow[j] * scale8[j] + spsum[bufi][m];
        }

        v16bf pA0, pA1;
        {
            const __bf16* pr0 = pbuf[bufi] + ln * 64 + hi * 16; // keys kb..+31
            v8bf lo = *(const v8bf*)(pr0);
            v8bf h8 = *(const v8bf*)(pr0 + 8);
            #pragma unroll
            for (int i = 0; i < 8; i++) { pA0[i] = lo[i]; pA0[i + 8] = h8[i]; }
            const __bf16* pr1 = pr0 + 32;                      // keys kb+32..+63
            v8bf lo1 = *(const v8bf*)(pr1);
            v8bf h81 = *(const v8bf*)(pr1 + 8);
            #pragma unroll
            for (int i = 0; i < 8; i++) { pA1[i] = lo1[i]; pA1[i + 8] = h81[i]; }
        }

        if (kb + 64 < NPIX)
            __builtin_prefetch(hT_b + (size_t)(c0 + ln) * NPIX + kb + 64, 0, 1);

        #pragma unroll
        for (int t = 0; t < 8; t++) {
            #pragma unroll
            for (int j = 0; j < 8; j++) acc[t][j] *= scale8[j];
            const __bf16* hp =
                hT_b + (size_t)(c0 + 16 * t + ln) * NPIX + kb + hi * 16;
            v16bf hB0 = *(const v16bf*)hp;
            acc[t] = __builtin_amdgcn_wmma_f32_16x16x32_bf16(
                         false, pA0, false, hB0, (short)0, acc[t], false, false);
            v16bf hB1 = *(const v16bf*)(hp + 32);
            acc[t] = __builtin_amdgcn_wmma_f32_16x16x32_bf16(
                         false, pA1, false, hB1, (short)0, acc[t], false, false);
        }
    }

    // ---- epilogue: out = gamma * (o / l) + x
    const float gm = gamma[0];
    #pragma unroll
    for (int t = 0; t < 8; t++) {
        #pragma unroll
        for (int j = 0; j < 8; j++) {
            int m = j + 8 * hi;
            size_t idx = (rowbase + q0 + m) * CCH + c0 + 16 * t + ln;
            out[idx] = gm * (acc[t][j] / lrow[j]) + x[idx];
        }
    }
}

// ---------------------------------------------------------------------------
extern "C" void kernel_launch(void* const* d_in, const int* in_sizes, int n_in,
                              void* d_out, int out_size, void* d_ws, size_t ws_size,
                              hipStream_t stream) {
    const float* x     = (const float*)d_in[0];
    const float* wf    = (const float*)d_in[1];
    const float* wg    = (const float*)d_in[2];
    const float* wh    = (const float*)d_in[3];
    const float* bf    = (const float*)d_in[4];
    const float* bg    = (const float*)d_in[5];
    const float* bh    = (const float*)d_in[6];
    const float* gamma = (const float*)d_in[7];
    float* out = (float*)d_out;

    // workspace: f [BN][32] bf16, g [BN][32] bf16, hT [B][C][N] bf16 (20 MB)
    __bf16* fbuf = (__bf16*)d_ws;
    __bf16* gbuf = fbuf + (size_t)BN * CFP;
    __bf16* hT   = gbuf + (size_t)BN * CFP;

    dim3 pgrid(BN / 16, 20);            // 2048 row tiles x (2 f + 2 g + 16 h)
    proj_kernel<<<pgrid, dim3(32), 0, stream>>>(
        x, wf, wg, wh, bf, bg, bh, fbuf, gbuf, hT);

    dim3 agrid(NPIX / 16, BATCH);       // 256 query tiles x 8 batches
    attn_kernel<<<agrid, dim3(64), 0, stream>>>(
        x, fbuf, gbuf, hT, gamma, out);
}
